// DiTAttention_55422257987618
// MI455X (gfx1250) — compile-verified, matrix operand order checked by reading, not run on hardware
//
#include <hip/hip_runtime.h>

typedef __attribute__((ext_vector_type(16))) __bf16 v16bf;
typedef __attribute__((ext_vector_type(8)))  float  v8f;
typedef unsigned short u16;

__device__ __forceinline__ u16 f2b(float f) {
    unsigned int u = __float_as_uint(f);
    unsigned int r = u + 0x7FFFu + ((u >> 16) & 1u);   // round-to-nearest-even
    return (u16)(r >> 16);
}
__device__ __forceinline__ __bf16 b2h(u16 s) {
    return __builtin_bit_cast(__bf16, s);
}

union U4 { uint4 v; u16 s[8]; };

// ---------------------------------------------------------------------------
// Kernel 1: fp32 -> bf16 for Wqkv (1024x3072), Wout (1024x1024), x (8192x1024)
// Memory bound; VALU here is free.
// ---------------------------------------------------------------------------
__global__ __launch_bounds__(256) void cvt_w(
    const float* __restrict__ Wqkv, const float* __restrict__ Wout,
    const float* __restrict__ X,
    u16* __restrict__ Wqkv_b, u16* __restrict__ Wout_b, u16* __restrict__ Xb)
{
    size_t i = (size_t)blockIdx.x * 256 + threadIdx.x;
    const size_t nq = (size_t)1024 * 3072;
    const size_t no = (size_t)1024 * 1024;
    const size_t nx = (size_t)8192 * 1024;
    if (i < nq) {
        Wqkv_b[i] = f2b(Wqkv[i]);
    } else if (i < nq + no) {
        size_t j = i - nq;
        Wout_b[j] = f2b(Wout[j]);
    } else if (i < nq + no + nx) {
        size_t j = i - nq - no;
        Xb[j] = f2b(X[j]);
    }
}

// ---------------------------------------------------------------------------
// Kernel 2: QKV GEMM (8192x1024 @ 1024x3072) + fused bias + RMSNorm + RoPE.
// 256 threads = 8 waves. Tile 128(M) x 64(N), K-step 32.
// A is pre-converted bf16; B stored transposed in LDS so every fragment
// gather is 2 x ds_load_b128.
// ---------------------------------------------------------------------------
__global__ __launch_bounds__(256) void qkv_gemm(
    const u16* __restrict__ Xb, const u16* __restrict__ Wb,
    const float* __restrict__ bqkv, const float* __restrict__ qnw,
    const float* __restrict__ knw,
    u16* __restrict__ qb, u16* __restrict__ kb, u16* __restrict__ vb)
{
    const int n0   = blockIdx.x * 64;    // 0..3071
    const int m0   = blockIdx.y * 128;   // 0..8191
    const int tid  = threadIdx.x;
    const int wave = tid >> 5, lane = tid & 31;
    const int lm   = lane & 15, hs = lane >> 4;
    const int bk   = hs * 8;

    __shared__ u16 sA[128][40];   // 128 x 32 bf16, row-major   (+pad)
    __shared__ u16 sBT[64][40];   //  64(n) x 32(k) transposed  (+pad)

    v8f acc[4] = {};

    for (int kk = 0; kk < 1024; kk += 32) {
        if (kk + 32 < 1024) {      // branch-free prefetch of next K-step
            __builtin_prefetch(Xb + (size_t)(m0 + (tid & 127)) * 1024 + kk + 32, 0, 3);
            __builtin_prefetch(Wb + (size_t)(kk + 32 + (tid & 31)) * 3072 + n0, 0, 3);
        }
        __syncthreads();
        // A: 128x32 bf16 (512 uint4, 2 per thread)
        #pragma unroll
        for (int i = 0; i < 2; ++i) {
            int f = tid + i * 256;
            int r = f >> 2, c8 = (f & 3) << 3;
            *reinterpret_cast<uint4*>(&sA[r][c8]) =
                *reinterpret_cast<const uint4*>(Xb + (size_t)(m0 + r) * 1024 + kk + c8);
        }
        // B: 32(k) x 64(n) -> transposed into sBT[n][k], two k's per b32 store
        if (tid < 128) {
            int rp = tid >> 3;          // k row-pair 0..15
            int cg = (tid & 7) << 3;    // n col group
            U4 u0, u1;
            u0.v = *reinterpret_cast<const uint4*>(Wb + (size_t)(kk + 2 * rp + 0) * 3072 + n0 + cg);
            u1.v = *reinterpret_cast<const uint4*>(Wb + (size_t)(kk + 2 * rp + 1) * 3072 + n0 + cg);
            #pragma unroll
            for (int j = 0; j < 8; ++j) {
                *reinterpret_cast<unsigned int*>(&sBT[cg + j][2 * rp]) =
                    (unsigned int)u0.s[j] | ((unsigned int)u1.s[j] << 16);
            }
        }
        __syncthreads();

        v16bf a;
        #pragma unroll
        for (int e = 0; e < 8; ++e) {
            a[e]     = b2h(sA[wave * 16 + lm][bk + e]);
            a[e + 8] = b2h(sA[wave * 16 + lm][16 + bk + e]);
        }
        v16bf bfr[4];
        #pragma unroll
        for (int n = 0; n < 4; ++n) {
            #pragma unroll
            for (int e = 0; e < 8; ++e) {
                bfr[n][e]     = b2h(sBT[n * 16 + lm][bk + e]);
                bfr[n][e + 8] = b2h(sBT[n * 16 + lm][16 + bk + e]);
            }
        }
        #pragma unroll
        for (int n = 0; n < 4; ++n)
            acc[n] = __builtin_amdgcn_wmma_f32_16x16x32_bf16(
                false, a, false, bfr[n], (short)0, acc[n], false, false);
    }

    const int t = n0 >> 10;          // 0=q 1=k 2=v
    const int h = (n0 & 1023) >> 6;  // head

    #pragma unroll
    for (int n = 0; n < 4; ++n) {
        float bias = bqkv[n0 + n * 16 + lm];
        #pragma unroll
        for (int v = 0; v < 8; ++v) acc[n][v] += bias;
    }

    if (t < 2) {
        const float* wnorm = (t == 0) ? qnw : knw;
        // RMSNorm over HD=64 (row = 16 lanes x 4 subtiles)
        #pragma unroll
        for (int v = 0; v < 8; ++v) {
            float ss = 0.f;
            #pragma unroll
            for (int n = 0; n < 4; ++n) ss += acc[n][v] * acc[n][v];
            ss += __shfl_xor(ss, 1, 32);
            ss += __shfl_xor(ss, 2, 32);
            ss += __shfl_xor(ss, 4, 32);
            ss += __shfl_xor(ss, 8, 32);
            float r = rsqrtf(ss * (1.0f / 64.0f) + 1e-6f);
            #pragma unroll
            for (int n = 0; n < 4; ++n) acc[n][v] *= r * wnorm[n * 16 + lm];
        }
        // RoPE: pair (c, c+32); c = j*16+lm in subtile j, c+32 in j+2 (same lane)
        const int rowbase = m0 + wave * 16 + hs * 8;
        #pragma unroll
        for (int v = 0; v < 8; ++v) {
            float pos = (float)((rowbase + v) & 2047);
            #pragma unroll
            for (int j = 0; j < 2; ++j) {
                int c = j * 16 + lm;
                float ang = pos * __expf(-0.28782314f * (float)c); // 10000^(-c/32)
                float sn, cs;
                __sincosf(ang, &sn, &cs);
                float t1 = acc[j][v], t2 = acc[j + 2][v];
                acc[j][v]     = t1 * cs - t2 * sn;
                acc[j + 2][v] = t1 * sn + t2 * cs;
            }
        }
    }

    // store bf16 into (B,H,L,HD)
    u16* dst = (t == 0) ? qb : ((t == 1) ? kb : vb);
    #pragma unroll
    for (int v = 0; v < 8; ++v) {
        int row = m0 + wave * 16 + hs * 8 + v;      // b*2048 + l
        int bb = row >> 11, ll = row & 2047;
        size_t base = (((size_t)(bb * 16 + h) * 2048) + ll) * 64;
        #pragma unroll
        for (int n = 0; n < 4; ++n)
            dst[base + n * 16 + lm] = f2b(acc[n][v]);
    }
}

// ---------------------------------------------------------------------------
// Kernel 3: flash attention. grid = (L/128, B*H), 256 threads = 8 waves.
// Each wave owns 16 query rows; loop over 32 key tiles of 64.
// V stored transposed (sVT[d][key]); K row-major. Fragment fetches batched
// in pairs to overlap DS traffic with the WMMA pipe at bounded VGPR cost.
// ---------------------------------------------------------------------------
__global__ __launch_bounds__(256) void attn(
    const u16* __restrict__ qb, const u16* __restrict__ kbuf,
    const u16* __restrict__ vbuf, u16* __restrict__ ob)
{
    const int bh = blockIdx.y;
    const int b = bh >> 4, h = bh & 15;
    const int q0 = blockIdx.x * 128;
    const int tid = threadIdx.x, wave = tid >> 5, lane = tid & 31;
    const int lm = lane & 15, hs = lane >> 4, bk = hs * 8;
    const size_t head_base = (size_t)bh * 2048 * 64;

    __shared__ u16 sQ[128][72];
    __shared__ u16 sK[64][72];      // [key][d]
    __shared__ u16 sVT[64][72];     // [d][key]  (transposed)
    __shared__ u16 sP[8][16][72];   // per-wave P tile [qrow][key]

    // Q tile -> LDS (1024 uint4, 4 per thread)
    #pragma unroll
    for (int i = 0; i < 4; ++i) {
        int f = tid + i * 256;
        int r = f >> 3, c8 = (f & 7) << 3;
        *reinterpret_cast<uint4*>(&sQ[r][c8]) =
            *reinterpret_cast<const uint4*>(qb + head_base + (size_t)(q0 + r) * 64 + c8);
    }
    __syncthreads();

    v16bf qa0, qa1;   // A-fragments of Q (K-dim = d: 0-31, 32-63)
    #pragma unroll
    for (int e = 0; e < 8; ++e) {
        qa0[e]     = b2h(sQ[wave * 16 + lm][bk + e]);
        qa0[e + 8] = b2h(sQ[wave * 16 + lm][16 + bk + e]);
        qa1[e]     = b2h(sQ[wave * 16 + lm][32 + bk + e]);
        qa1[e + 8] = b2h(sQ[wave * 16 + lm][48 + bk + e]);
    }

    float m[8], l[8];
    v8f o[4] = {};
    #pragma unroll
    for (int v = 0; v < 8; ++v) { m[v] = -1e30f; l[v] = 0.f; }

    for (int kt = 0; kt < 32; ++kt) {
        const size_t tile = head_base + (size_t)kt * 64 * 64;
        if (kt + 1 < 32) {   // branch-free prefetch of next K/V tile
            __builtin_prefetch(kbuf + tile + 4096 + tid * 16, 0, 3);
            __builtin_prefetch(vbuf + tile + 4096 + tid * 16, 0, 3);
        }
        __syncthreads();
        // K: row-major (512 uint4, 2 per thread)
        #pragma unroll
        for (int i = 0; i < 2; ++i) {
            int f = tid + i * 256;
            int r = f >> 3, c8 = (f & 7) << 3;
            *reinterpret_cast<uint4*>(&sK[r][c8]) =
                *reinterpret_cast<const uint4*>(kbuf + tile + (size_t)r * 64 + c8);
        }
        // V: transpose into sVT[d][key], two keys per b32 (256 jobs)
        {
            int kp = tid >> 3;          // key pair 0..31
            int cg = (tid & 7) << 3;    // d col group
            U4 u0, u1;
            u0.v = *reinterpret_cast<const uint4*>(vbuf + tile + (size_t)(2 * kp + 0) * 64 + cg);
            u1.v = *reinterpret_cast<const uint4*>(vbuf + tile + (size_t)(2 * kp + 1) * 64 + cg);
            #pragma unroll
            for (int j = 0; j < 8; ++j) {
                *reinterpret_cast<unsigned int*>(&sVT[cg + j][2 * kp]) =
                    (unsigned int)u0.s[j] | ((unsigned int)u1.s[j] << 16);
            }
        }
        __syncthreads();

        // S = Q K^T (chained WMMA over d = 0-31, 32-63), key subtiles in pairs
        v8f s[4];
        #pragma unroll
        for (int np = 0; np < 2; ++np) {
            v16bf kf[4];   // {k0,k1} for n=2np, n=2np+1
            #pragma unroll
            for (int q = 0; q < 2; ++q) {
                int key = (2 * np + q) * 16 + lm;
                #pragma unroll
                for (int e = 0; e < 8; ++e) {
                    kf[2 * q + 0][e]     = b2h(sK[key][bk + e]);
                    kf[2 * q + 0][e + 8] = b2h(sK[key][16 + bk + e]);
                    kf[2 * q + 1][e]     = b2h(sK[key][32 + bk + e]);
                    kf[2 * q + 1][e + 8] = b2h(sK[key][48 + bk + e]);
                }
            }
            #pragma unroll
            for (int q = 0; q < 2; ++q) {
                v8f z = {};
                z = __builtin_amdgcn_wmma_f32_16x16x32_bf16(false, qa0, false, kf[2 * q + 0], (short)0, z, false, false);
                z = __builtin_amdgcn_wmma_f32_16x16x32_bf16(false, qa1, false, kf[2 * q + 1], (short)0, z, false, false);
                s[2 * np + q] = z;
            }
        }

        // online softmax; scale = HD^-0.5 = 0.125
        #pragma unroll
        for (int v = 0; v < 8; ++v) {
            float rm = -1e30f;
            #pragma unroll
            for (int n = 0; n < 4; ++n) { s[n][v] *= 0.125f; rm = fmaxf(rm, s[n][v]); }
            rm = fmaxf(rm, __shfl_xor(rm, 1, 32));
            rm = fmaxf(rm, __shfl_xor(rm, 2, 32));
            rm = fmaxf(rm, __shfl_xor(rm, 4, 32));
            rm = fmaxf(rm, __shfl_xor(rm, 8, 32));
            float mn = fmaxf(m[v], rm);
            float alpha = __expf(m[v] - mn);
            float rs = 0.f;
            #pragma unroll
            for (int n = 0; n < 4; ++n) {
                float p = __expf(s[n][v] - mn);
                rs += p;
                sP[wave][v + hs * 8][n * 16 + lm] = f2b(p);
            }
            rs += __shfl_xor(rs, 1, 32);
            rs += __shfl_xor(rs, 2, 32);
            rs += __shfl_xor(rs, 4, 32);
            rs += __shfl_xor(rs, 8, 32);
            l[v] = l[v] * alpha + rs;
            m[v] = mn;
            #pragma unroll
            for (int j = 0; j < 4; ++j) o[j][v] *= alpha;
        }
        // wave-local LDS RAW (C-layout -> A-layout repack); DS in-order per wave
        asm volatile("s_wait_dscnt 0" ::: "memory");

        // O += P V, d subtiles in pairs
        v16bf pa0, pa1;
        #pragma unroll
        for (int e = 0; e < 8; ++e) {
            pa0[e]     = b2h(sP[wave][lm][bk + e]);
            pa0[e + 8] = b2h(sP[wave][lm][16 + bk + e]);
            pa1[e]     = b2h(sP[wave][lm][32 + bk + e]);
            pa1[e + 8] = b2h(sP[wave][lm][48 + bk + e]);
        }
        #pragma unroll
        for (int jp = 0; jp < 2; ++jp) {
            v16bf vf[4];   // {v0,v1} for j=2jp, j=2jp+1
            #pragma unroll
            for (int q = 0; q < 2; ++q) {
                int d = (2 * jp + q) * 16 + lm;
                #pragma unroll
                for (int e = 0; e < 8; ++e) {
                    vf[2 * q + 0][e]     = b2h(sVT[d][bk + e]);
                    vf[2 * q + 0][e + 8] = b2h(sVT[d][16 + bk + e]);
                    vf[2 * q + 1][e]     = b2h(sVT[d][32 + bk + e]);
                    vf[2 * q + 1][e + 8] = b2h(sVT[d][48 + bk + e]);
                }
            }
            #pragma unroll
            for (int q = 0; q < 2; ++q) {
                int j = 2 * jp + q;
                o[j] = __builtin_amdgcn_wmma_f32_16x16x32_bf16(false, pa0, false, vf[2 * q + 0], (short)0, o[j], false, false);
                o[j] = __builtin_amdgcn_wmma_f32_16x16x32_bf16(false, pa1, false, vf[2 * q + 1], (short)0, o[j], false, false);
            }
        }
    }

    // normalize and store bf16 directly in (B, L, H*HD) = (B,L,D)
    #pragma unroll
    for (int v = 0; v < 8; ++v) {
        float inv = 1.0f / l[v];
        int grow = q0 + wave * 16 + hs * 8 + v;
        size_t base = ((size_t)b * 2048 + grow) * 1024 + h * 64;
        #pragma unroll
        for (int j = 0; j < 4; ++j)
            ob[base + j * 16 + lm] = f2b(o[j][v] * inv);
    }
}

// ---------------------------------------------------------------------------
// Kernel 4: out GEMM (8192x1024 bf16 @ 1024x1024 bf16) + bias -> f32
// 256 threads = 8 waves, tile 128 x 64, K-step 32, transposed-B LDS.
// ---------------------------------------------------------------------------
__global__ __launch_bounds__(256) void out_gemm(
    const u16* __restrict__ A, const u16* __restrict__ Wb,
    const float* __restrict__ bout, float* __restrict__ out)
{
    const int n0 = blockIdx.x * 64;
    const int m0 = blockIdx.y * 128;
    const int tid = threadIdx.x;
    const int wave = tid >> 5, lane = tid & 31;
    const int lm = lane & 15, hs = lane >> 4, bk = hs * 8;

    __shared__ u16 sA[128][40];
    __shared__ u16 sBT[64][40];

    v8f acc[4] = {};

    for (int kk = 0; kk < 1024; kk += 32) {
        if (kk + 32 < 1024) {      // branch-free prefetch of next K-step
            __builtin_prefetch(A + (size_t)(m0 + (tid & 127)) * 1024 + kk + 32, 0, 3);
            __builtin_prefetch(Wb + (size_t)(kk + 32 + (tid & 31)) * 1024 + n0, 0, 3);
        }
        __syncthreads();
        // A: 128x32 bf16 (512 uint4, 2 per thread)
        #pragma unroll
        for (int i = 0; i < 2; ++i) {
            int f = tid + i * 256;
            int r = f >> 2, c8 = (f & 3) << 3;
            *reinterpret_cast<uint4*>(&sA[r][c8]) =
                *reinterpret_cast<const uint4*>(A + (size_t)(m0 + r) * 1024 + kk + c8);
        }
        // B transposed into sBT[n][k]
        if (tid < 128) {
            int rp = tid >> 3;
            int cg = (tid & 7) << 3;
            U4 u0, u1;
            u0.v = *reinterpret_cast<const uint4*>(Wb + (size_t)(kk + 2 * rp + 0) * 1024 + n0 + cg);
            u1.v = *reinterpret_cast<const uint4*>(Wb + (size_t)(kk + 2 * rp + 1) * 1024 + n0 + cg);
            #pragma unroll
            for (int j = 0; j < 8; ++j) {
                *reinterpret_cast<unsigned int*>(&sBT[cg + j][2 * rp]) =
                    (unsigned int)u0.s[j] | ((unsigned int)u1.s[j] << 16);
            }
        }
        __syncthreads();

        v16bf a;
        #pragma unroll
        for (int e = 0; e < 8; ++e) {
            a[e]     = b2h(sA[wave * 16 + lm][bk + e]);
            a[e + 8] = b2h(sA[wave * 16 + lm][16 + bk + e]);
        }
        v16bf bfr[4];
        #pragma unroll
        for (int n = 0; n < 4; ++n) {
            #pragma unroll
            for (int e = 0; e < 8; ++e) {
                bfr[n][e]     = b2h(sBT[n * 16 + lm][bk + e]);
                bfr[n][e + 8] = b2h(sBT[n * 16 + lm][16 + bk + e]);
            }
        }
        #pragma unroll
        for (int n = 0; n < 4; ++n)
            acc[n] = __builtin_amdgcn_wmma_f32_16x16x32_bf16(
                false, a, false, bfr[n], (short)0, acc[n], false, false);
    }

    #pragma unroll
    for (int v = 0; v < 8; ++v) {
        int row = m0 + wave * 16 + hs * 8 + v;
        #pragma unroll
        for (int n = 0; n < 4; ++n) {
            int col = n0 + n * 16 + lm;
            out[(size_t)row * 1024 + col] = acc[n][v] + bout[col];
        }
    }
}

// ---------------------------------------------------------------------------
extern "C" void kernel_launch(void* const* d_in, const int* in_sizes, int n_in,
                              void* d_out, int out_size, void* d_ws, size_t ws_size,
                              hipStream_t stream) {
    const float* x    = (const float*)d_in[0];
    const float* Wqkv = (const float*)d_in[1];
    const float* bqkv = (const float*)d_in[2];
    const float* qnw  = (const float*)d_in[3];
    const float* knw  = (const float*)d_in[4];
    const float* Wout = (const float*)d_in[5];
    const float* bout = (const float*)d_in[6];
    float* out = (float*)d_out;

    char* ws = (char*)d_ws;
    u16* Wqkv_b = (u16*)(ws);                    //  6 MB
    u16* Wout_b = (u16*)(ws + 6291456);          //  2 MB
    u16* xbuf   = (u16*)(ws + 8388608);          // 16 MB (x bf16)
    u16* qbuf   = (u16*)(ws + 25165824);         // 16 MB
    u16* kbuf   = (u16*)(ws + 41943040);         // 16 MB
    u16* vbuf   = (u16*)(ws + 58720256);         // 16 MB
    u16* abuf   = xbuf;                          // reuse: x dead after qkv_gemm

    cvt_w<<<49152, 256, 0, stream>>>(Wqkv, Wout, x, Wqkv_b, Wout_b, xbuf);
    qkv_gemm<<<dim3(48, 64), 256, 0, stream>>>(xbuf, Wqkv_b, bqkv, qnw, knw,
                                               qbuf, kbuf, vbuf);
    attn<<<dim3(16, 64), 256, 0, stream>>>(qbuf, kbuf, vbuf, abuf);
    out_gemm<<<dim3(16, 64), 256, 0, stream>>>(abuf, Wout_b, bout, out);
}
